// PairContrastiveLoss_69174743269978
// MI455X (gfx1250) — matrix operand order repaired and checked by reading, not compiled, not run on hardware
//
#include <hip/hip_runtime.h>
#include <hip/hip_bf16.h>
#include <math.h>

// ---------------------------------------------------------------------------
// PairContrastiveLoss on MI455X (gfx1250, wave32, WMMA bf16, async G->LDS)
//
//  x,y [32,3,224,224] --im2row--> A [6272,768] bf16
//  tok = A @ convW^T + bias + pos_emb  (WMMA GEMM, epilogue fused)
//  sim = xv @ yv^T / max(|x||y|,eps); e = exp(2*sim)
//  loss = log((sum(e)-trace(e)) / trace(e))
//
//  Token buffers padded to 6400 rows (zeroed), norm arrays padded with 1.0,
//  so all staging copies and epilogue norm reads are unconditional.
// ---------------------------------------------------------------------------

typedef __bf16  v16bf __attribute__((ext_vector_type(16)));
typedef __bf16  v8bf  __attribute__((ext_vector_type(8)));
typedef float   v8f   __attribute__((ext_vector_type(8)));

#define HIDDEN   768
#define NB       32
#define NPATCH   196
#define NTOK     197
#define NTOT     (NB * NTOK)    /* 6304 */
#define NPADROWS 6400           /* 50 * 128, >= 99*64 = 6336 too */
#define MROWS    (NB * NPATCH)  /* 6272 */

#define LDSA     40             /* padded K-stride in bf16 (80B, 16B aligned) */

// ---------------- CDNA5 async global -> LDS helpers ----------------
__device__ __forceinline__ void async_g2l_b128(void* lds_ptr, const void* gptr) {
    // VDST = LDS byte offset (low 32 bits of generic shared address),
    // VADDR = 64-bit global address. Tracked by ASYNCcnt.
    unsigned lds_off = (unsigned)(size_t)lds_ptr;
    unsigned long long ga = (unsigned long long)(size_t)gptr;
    asm volatile("global_load_async_to_lds_b128 %0, %1, off"
                 :: "v"(lds_off), "v"(ga)
                 : "memory");
}

__device__ __forceinline__ void wait_async0() {
#if __has_builtin(__builtin_amdgcn_s_wait_asynccnt)
    __builtin_amdgcn_s_wait_asynccnt(0);
#else
    asm volatile("s_wait_asynccnt 0x0" ::: "memory");
#endif
}

// ---------------------------------------------------------------------------
__global__ void pcl_init(float* sums) {
    if (threadIdx.x < 2) sums[threadIdx.x] = 0.0f;
}

// zero the pad rows [6304, 6400) of both token buffers (as u32 words),
// and set the pad entries of both norm arrays to 1.0
__global__ void pcl_pad(__bf16* xt, __bf16* yt, float* xn, float* yn) {
    const int words = (NPADROWS - NTOT) * HIDDEN / 2;      // 36864 per buffer
    int idx = blockIdx.x * 256 + threadIdx.x;              // < 2*words
    unsigned* xz = (unsigned*)(xt + (size_t)NTOT * HIDDEN);
    unsigned* yz = (unsigned*)(yt + (size_t)NTOT * HIDDEN);
    if (idx < words) xz[idx] = 0u;
    else             yz[idx - words] = 0u;
    if (idx < NPADROWS - NTOT) {                           // 96 pad norms
        xn[NTOT + idx] = 1.0f;
        yn[NTOT + idx] = 1.0f;
    }
}

// conv_w [768,768] f32 row-major (N,K) -> bf16 copy (already the B^T layout)
__global__ void pcl_cvt_w(const float* __restrict__ w, __bf16* __restrict__ wb) {
    int i = blockIdx.x * 256 + threadIdx.x;
    wb[i] = (__bf16)w[i];
}

// img [32,3,224,224] -> A [6272,768] bf16, k = c*256 + r*16 + cc
__global__ void pcl_im2row(const float* __restrict__ img, __bf16* __restrict__ out) {
    size_t idx = (size_t)blockIdx.x * 256 + threadIdx.x;   // < 6272*768
    int kk  = (int)(idx % HIDDEN);
    int row = (int)(idx / HIDDEN);
    int b  = row / NPATCH, p = row % NPATCH;
    int ph = p / 14,       pw = p % 14;
    int c  = kk >> 8, r = (kk >> 4) & 15, cc = kk & 15;
    size_t src = ((size_t)(b * 3 + c) * 224 + (ph * 16 + r)) * 224 + (pw * 16 + cc);
    out[idx] = (__bf16)img[src];
}

// cls rows: tok[b*197, :] = cls + pos_emb[0, :] (identical for x and y)
__global__ void pcl_cls(const float* __restrict__ cls, const float* __restrict__ pos,
                        __bf16* __restrict__ xt, __bf16* __restrict__ yt) {
    int idx = blockIdx.x * 256 + threadIdx.x;      // < 32*768
    int b = idx / HIDDEN, d = idx % HIDDEN;
    __bf16 v = (__bf16)(cls[d] + pos[d]);
    size_t o = (size_t)(b * NTOK) * HIDDEN + d;
    xt[o] = v;
    yt[o] = v;
}

// one block per row (x rows then y rows): L2 norm
__global__ void pcl_rownorm(const __bf16* __restrict__ xt, const __bf16* __restrict__ yt,
                            float* __restrict__ xn, float* __restrict__ yn) {
    __shared__ float red[256];
    int row = blockIdx.x;
    const __bf16* t; float* o; int r;
    if (row < NTOT) { t = xt; o = xn; r = row; }
    else            { t = yt; o = yn; r = row - NTOT; }
    float s = 0.0f;
    for (int i = threadIdx.x; i < HIDDEN; i += 256) {
        float v = (float)t[(size_t)r * HIDDEN + i];
        s += v * v;
    }
    red[threadIdx.x] = s; __syncthreads();
    for (int st = 128; st > 0; st >>= 1) {
        if (threadIdx.x < st) red[threadIdx.x] += red[threadIdx.x + st];
        __syncthreads();
    }
    if (threadIdx.x == 0) o[r] = sqrtf(red[0]);
}

__global__ void pcl_final(const float* __restrict__ sums, float* __restrict__ out) {
    float tot = sums[0], diag = sums[1];
    out[0] = logf((tot - diag) / diag);    // -log(close/far)
}

// ---------------------------------------------------------------------------
// Unified bf16 WMMA GEMM:  C[M,N] = A[M,K=768] * Bt[N,K=768]^T
//   mode 0: embed epilogue (bias + pos_emb, write bf16 tokens)
//   mode 1: similarity epilogue (cosine, exp(2*sim), block reduce + atomics)
// Block tile 128x64, BK=32, 256 threads = 8 waves (4 M x 2 N), wave = 32x32.
// Double-buffered LDS, staged with global_load_async_to_lds_b128 (ASYNCcnt).
// A and Bt buffers must be readable for the full padded tile extents.
// ---------------------------------------------------------------------------
__global__ __launch_bounds__(256)
void pcl_gemm_wmma(const __bf16* __restrict__ A, const __bf16* __restrict__ Bt,
                   int M, int Nn, int mode,
                   const float* __restrict__ convb, const float* __restrict__ posemb,
                   __bf16* __restrict__ tokout,
                   const float* __restrict__ xn, const float* __restrict__ yn,
                   float* __restrict__ sums) {
    __shared__ __align__(16) __bf16 As[2][128 * LDSA];
    __shared__ __align__(16) __bf16 Bs[2][64  * LDSA];
    __shared__ float red[256];
    __shared__ float xns[128];
    __shared__ float yns[64];

    const int tid  = threadIdx.x;
    const int lane = tid & 31;
    const int wid  = tid >> 5;
    const int wm   = wid & 3;      // wave M index (4 x 32 = 128)
    const int wn   = wid >> 2;     // wave N index (2 x 32 = 64)
    const int rowBase = blockIdx.x * 128;
    const int colBase = blockIdx.y * 64;

    const int half = lane >> 4;    // 0/1: which 16-lane group
    const int ml   = lane & 15;

    // per-thread staging coordinates (A: 2 x 16B segs, B: 1 x 16B seg)
    const int ar0 = tid >> 2,          ac0 = (tid & 3) * 8;
    const int ar1 = (tid + 256) >> 2,  ac1 = ac0;
    const int br  = tid >> 2,          bc  = (tid & 3) * 8;
    const __bf16* agp0 = A  + (size_t)(rowBase + ar0) * HIDDEN + ac0;
    const __bf16* agp1 = A  + (size_t)(rowBase + ar1) * HIDDEN + ac1;
    const __bf16* bgp  = Bt + (size_t)(colBase + br)  * HIDDEN + bc;

    v8f acc[2][2] = {};

    // prologue: stage tile 0 into buffer 0
    async_g2l_b128(&As[0][ar0 * LDSA + ac0], agp0);
    async_g2l_b128(&As[0][ar1 * LDSA + ac1], agp1);
    async_g2l_b128(&Bs[0][br  * LDSA + bc ], bgp);

    const int KT = HIDDEN / 32;            // 24 K-tiles
    for (int it = 0; it < KT; ++it) {
        const int cur = it & 1;
        wait_async0();                     // my async writes for tile `it` done
        __syncthreads();                   // everyone's writes done; prev reads done

        if (it + 1 < KT) {                 // stage next tile into other buffer
            const int kb = (it + 1) * 32;
            async_g2l_b128(&As[cur ^ 1][ar0 * LDSA + ac0], agp0 + kb);
            async_g2l_b128(&As[cur ^ 1][ar1 * LDSA + ac1], agp1 + kb);
            async_g2l_b128(&Bs[cur ^ 1][br  * LDSA + bc ], bgp  + kb);
        }

        // ---- build fragments per documented CDNA5 VGPR layouts ----
        // A 16x32 bf16: lane m = lane%16; K = half*8..+7 and 16+half*8..+7
        v16bf af[2], bfr[2];
#pragma unroll
        for (int mi = 0; mi < 2; ++mi) {
            const __bf16* p = &As[cur][(wm * 32 + mi * 16 + ml) * LDSA];
            v8bf lo = *(const v8bf*)(p + half * 8);
            v8bf hi = *(const v8bf*)(p + 16 + half * 8);
            af[mi] = __builtin_shufflevector(lo, hi, 0,1,2,3,4,5,6,7,8,9,10,11,12,13,14,15);
        }
        // B 32x16 bf16: lane n = lane%16; K = half*8..+7 and 16+half*8..+7
#pragma unroll
        for (int ni = 0; ni < 2; ++ni) {
            const __bf16* p = &Bs[cur][(wn * 32 + ni * 16 + ml) * LDSA];
            v8bf lo = *(const v8bf*)(p + half * 8);
            v8bf hi = *(const v8bf*)(p + 16 + half * 8);
            bfr[ni] = __builtin_shufflevector(lo, hi, 0,1,2,3,4,5,6,7,8,9,10,11,12,13,14,15);
        }

#pragma unroll
        for (int mi = 0; mi < 2; ++mi)
#pragma unroll
            for (int ni = 0; ni < 2; ++ni)
                acc[mi][ni] = __builtin_amdgcn_wmma_f32_16x16x32_bf16(
                    false, af[mi], false, bfr[ni], (short)0, acc[mi][ni], false, false);
    }

    // ------------------------- epilogues -------------------------
    if (mode == 0) {
        // M=6272, Nn=768, grid exact -> no bounds needed
#pragma unroll
        for (int mi = 0; mi < 2; ++mi) {
#pragma unroll
            for (int ni = 0; ni < 2; ++ni) {
                int n = colBase + wn * 32 + ni * 16 + ml;
#pragma unroll
                for (int r = 0; r < 8; ++r) {
                    int m = rowBase + wm * 32 + mi * 16 + r + 8 * half;
                    int p    = m % NPATCH;
                    int b    = m / NPATCH;
                    int trow = b * NTOK + 1 + p;
                    float val = acc[mi][ni][r] + convb[n]
                              + posemb[(size_t)(1 + p) * HIDDEN + n];
                    tokout[(size_t)trow * HIDDEN + n] = (__bf16)val;
                }
            }
        }
    } else {
        // stage this block's norms into LDS (padded arrays -> unconditional)
        if (tid < 128)      xns[tid]       = xn[rowBase + tid];
        else if (tid < 192) yns[tid - 128] = yn[colBase + (tid - 128)];
        __syncthreads();

        float lsum = 0.0f, ldiag = 0.0f;
#pragma unroll
        for (int mi = 0; mi < 2; ++mi) {
#pragma unroll
            for (int ni = 0; ni < 2; ++ni) {
                int nl = wn * 32 + ni * 16 + ml;
                int n  = colBase + nl;
                float ynv = yns[nl];
#pragma unroll
                for (int r = 0; r < 8; ++r) {
                    int ml2 = wm * 32 + mi * 16 + r + 8 * half;
                    int m   = rowBase + ml2;
                    float denom = fmaxf(xns[ml2] * ynv, 1e-8f);
                    float e = __expf(2.0f * (acc[mi][ni][r] / denom)); // sim/TEMP
                    bool in = (m < M) && (n < Nn);
                    lsum += in ? e : 0.0f;
                    ldiag += (in && (m == n)) ? e : 0.0f;
                }
            }
        }

        red[tid] = lsum; __syncthreads();
        for (int s = 128; s > 0; s >>= 1) {
            if (tid < s) red[tid] += red[tid + s];
            __syncthreads();
        }
        if (tid == 0) atomicAdd(&sums[0], red[0]);
        __syncthreads();
        red[tid] = ldiag; __syncthreads();
        for (int s = 128; s > 0; s >>= 1) {
            if (tid < s) red[tid] += red[tid + s];
            __syncthreads();
        }
        if (tid == 0) atomicAdd(&sums[1], red[0]);
    }
}

// ---------------------------------------------------------------------------
extern "C" void kernel_launch(void* const* d_in, const int* in_sizes, int n_in,
                              void* d_out, int out_size, void* d_ws, size_t ws_size,
                              hipStream_t stream) {
    const float* x        = (const float*)d_in[0];
    const float* y        = (const float*)d_in[1];
    const float* conv_w   = (const float*)d_in[2];
    const float* conv_b   = (const float*)d_in[3];
    const float* pos_emb  = (const float*)d_in[4];
    const float* cls_tok  = (const float*)d_in[5];
    float* out = (float*)d_out;

    // workspace layout (all 16B aligned); token buffers padded to 6400 rows
    char* ws = (char*)d_ws;
    const size_t WTB   = (size_t)HIDDEN * HIDDEN * 2;        // 1,179,648
    const size_t ABUF  = (size_t)MROWS * HIDDEN * 2;         // 9,633,792
    const size_t TOK   = (size_t)NPADROWS * HIDDEN * 2;      // 9,830,400
    const size_t NRM   = (size_t)NPADROWS * 4;               // 25,600
    float*  sums = (float*)(ws);
    __bf16* wtb  = (__bf16*)(ws + 256);
    __bf16* abuf = (__bf16*)(ws + 256 + WTB);
    __bf16* xt   = (__bf16*)(ws + 256 + WTB + ABUF);
    __bf16* yt   = (__bf16*)(ws + 256 + WTB + ABUF + TOK);
    float*  xn   = (float*) (ws + 256 + WTB + ABUF + 2 * TOK);
    float*  yn   = (float*) (ws + 256 + WTB + ABUF + 2 * TOK + NRM);

    pcl_init<<<1, 32, 0, stream>>>(sums);
    pcl_pad<<<(2 * (NPADROWS - NTOT) * HIDDEN / 2) / 256, 256, 0, stream>>>(xt, yt, xn, yn);
    pcl_cvt_w<<<(HIDDEN * HIDDEN) / 256, 256, 0, stream>>>(conv_w, wtb);

    // x embed
    pcl_im2row<<<(MROWS * HIDDEN) / 256, 256, 0, stream>>>(x, abuf);
    pcl_gemm_wmma<<<dim3(MROWS / 128, HIDDEN / 64), 256, 0, stream>>>(
        abuf, wtb, MROWS, HIDDEN, 0, conv_b, pos_emb, xt, nullptr, nullptr, nullptr);

    // y embed (abuf reused; stream-ordered)
    pcl_im2row<<<(MROWS * HIDDEN) / 256, 256, 0, stream>>>(y, abuf);
    pcl_gemm_wmma<<<dim3(MROWS / 128, HIDDEN / 64), 256, 0, stream>>>(
        abuf, wtb, MROWS, HIDDEN, 0, conv_b, pos_emb, yt, nullptr, nullptr, nullptr);

    pcl_cls<<<(NB * HIDDEN) / 256, 256, 0, stream>>>(cls_tok, pos_emb, xt, yt);
    pcl_rownorm<<<2 * NTOT, 256, 0, stream>>>(xt, yt, xn, yn);

    // similarity GEMM + fused exp/reduce over padded row space
    pcl_gemm_wmma<<<dim3(NPADROWS / 128, (NTOT + 63) / 64), 256, 0, stream>>>(
        xt, yt, NTOT, NTOT, 1, nullptr, nullptr, nullptr, xn, yn, sums);

    pcl_final<<<1, 1, 0, stream>>>(sums, out);
}
